// MultiHeadAttention_8693013807740
// MI455X (gfx1250) — matrix operand order repaired and checked by reading, not compiled, not run on hardware
//
#include <hip/hip_runtime.h>

// ---------------------------------------------------------------------------
// MHA (B=4,S=1024,D=1024,H=8,L=256) on gfx1250: bf16 WMMA + async LDS staging
// ---------------------------------------------------------------------------

typedef __attribute__((ext_vector_type(16))) __bf16 v16bf;
typedef __attribute__((ext_vector_type(8)))  __bf16 v8bf;
typedef __attribute__((ext_vector_type(8)))  float  v8f;

#define B_  4
#define S_  1024
#define D_  1024
#define H_  8
#define L_  256
#define OQK (H_ * L_)   // 2048
#define OV  (H_ * D_)   // 8192
#define M_  (B_ * S_)   // 4096

#define LP_  (D_ + 8)   // padded LDS pitch (bf16 elems) for 1024-wide panels
#define LPQ_ (L_ + 8)   // padded LDS pitch for 256-wide q panel

// ---------------------------------------------------------------------------
// CDNA5 async global->LDS staging (guarded; falls back to load+ds_store)
// Builtin signature (from probe diagnostics): pointee is int vector_size(16),
// global operand in AS1 ("__device__"), LDS operand in AS3 ("__shared__").
// ---------------------------------------------------------------------------
#if __has_builtin(__builtin_amdgcn_global_load_async_to_lds_b128)
#define ASYNC_LDS 1
typedef int v4i_vs __attribute__((vector_size(4 * sizeof(int))));
typedef __attribute__((address_space(1))) v4i_vs gv4i;
typedef __attribute__((address_space(3))) v4i_vs lv4i;
#else
#define ASYNC_LDS 0
#endif

__device__ __forceinline__ void copy16_g2l(const __bf16* g, __bf16* l) {
#if ASYNC_LDS
  __builtin_amdgcn_global_load_async_to_lds_b128((gv4i*)g, (lv4i*)l, 0, 0);
#else
  *(int4*)l = *(const int4*)g;   // global_load_b128 + ds_store_b128
#endif
}

__device__ __forceinline__ void async_join() {
#if ASYNC_LDS
#if __has_builtin(__builtin_amdgcn_s_wait_asynccnt)
  __builtin_amdgcn_s_wait_asynccnt(0);
#else
  asm volatile("s_wait_asynccnt 0" ::: "memory");
#endif
#endif
  __syncthreads();
}

// float -> bf16, round-to-nearest-even
__device__ __forceinline__ __bf16 f2bf(float f) {
  unsigned u = __builtin_bit_cast(unsigned, f);
  u += 0x7FFFu + ((u >> 16) & 1u);
  unsigned short s = (unsigned short)(u >> 16);
  return __builtin_bit_cast(__bf16, s);
}

// A-operand 16x32 (row-major source, leading dim lda).
// A layout (contiguous, matches B): lane<16 row=lane holds K 0..15;
// lane>=16 row=lane-16 holds K 16..31.
__device__ __forceinline__ v16bf load_a16x32(const __bf16* tile, int lda, int lane) {
  int row  = lane & 15;
  int koff = (lane >> 4) << 4;               // 0 or 16
  const __bf16* p = tile + (size_t)row * lda + koff;
  v8bf lo = *(const v8bf*)p;
  v8bf hi = *(const v8bf*)(p + 8);
  v16bf a;
#pragma unroll
  for (int i = 0; i < 8; ++i) { a[i] = lo[i]; a[i + 8] = hi[i]; }
  return a;
}

// B-operand 32x16 where column n of B is CONTIGUOUS in memory (row pitch ldb):
// lane<16: col=lane, K 0..15 contiguous; lane>=16: col=lane-16, K 16..31.
__device__ __forceinline__ v16bf load_b32x16(const __bf16* tile, int ldb, int lane) {
  const __bf16* p = tile + (size_t)(lane & 15) * ldb + ((lane >> 4) << 4);
  return *(const v16bf*)p;
}

__device__ __forceinline__ v8f wmma_bf16(v16bf a, v16bf b, v8f c) {
  return __builtin_amdgcn_wmma_f32_16x16x32_bf16(false, a, false, b,
                                                 (short)0, c, false, false);
}

// ---------------------------------------------------------------------------
// 1) fp32 -> bf16 cast
// ---------------------------------------------------------------------------
__global__ void cast_f32_bf16(const float* __restrict__ in,
                              __bf16* __restrict__ out, int n) {
  int i = blockIdx.x * blockDim.x + threadIdx.x;
  int stride = gridDim.x * blockDim.x;
  for (; i < n; i += stride) out[i] = f2bf(in[i]);
}

// ---------------------------------------------------------------------------
// 2) C[M,O] = A[M,1024] @ W[O,1024]^T  (bf16 in, f32 accum, bf16 out)
//    block = 256 thr (8 waves); A panel staged once in LDS;
//    each wave computes a 16x32 tile (2 accumulators sharing one A operand)
// ---------------------------------------------------------------------------
__global__ void gemm_xwT(const __bf16* __restrict__ A,
                         const __bf16* __restrict__ W,
                         __bf16* __restrict__ C, int O) {
  __shared__ __align__(16) __bf16 lA[16 * LP_];
  int lane = threadIdx.x & 31;
  int wave = threadIdx.x >> 5;
  int m0 = blockIdx.x * 16;
  int o0 = blockIdx.y * 256 + wave * 32;

  // stage A panel: 16 rows x 1024 bf16 (32 KB), 16B per thread-iteration
  for (int idx = threadIdx.x; idx < 16 * (D_ / 8); idx += 256) {
    int row = idx >> 7;                 // idx / 128
    int c8  = (idx & 127) << 3;         // 8-elem chunk
    copy16_g2l(A + (size_t)(m0 + row) * D_ + c8, lA + row * LP_ + c8);
  }
  async_join();

  v8f acc0 = {}, acc1 = {};
  for (int k0 = 0; k0 < D_; k0 += 32) {
    __builtin_prefetch(W + (size_t)o0 * D_ + k0 + 256, 0, 0);
    v16bf a  = load_a16x32(lA + k0, LP_, lane);                     // ds_load
    v16bf b0 = load_b32x16(W + (size_t)o0 * D_ + k0, D_, lane);
    v16bf b1 = load_b32x16(W + (size_t)(o0 + 16) * D_ + k0, D_, lane);
    acc0 = wmma_bf16(a, b0, acc0);
    acc1 = wmma_bf16(a, b1, acc1);
  }
  int n    = o0 + (lane & 15);
  int mrow = m0 + ((lane >> 4) << 3);
#pragma unroll
  for (int r = 0; r < 8; ++r) {
    C[(size_t)(mrow + r) * O + n]      = f2bf(acc0[r]);
    C[(size_t)(mrow + r) * O + n + 16] = f2bf(acc1[r]);
  }
}

// ---------------------------------------------------------------------------
// 3) V projection with transposed store: VT[b][h][d][s]
// ---------------------------------------------------------------------------
struct alignas(16) BF8 { __bf16 v[8]; };

__global__ void gemm_vproj(const __bf16* __restrict__ A,   // [M, D]
                           const __bf16* __restrict__ W,   // [OV, D]
                           __bf16* __restrict__ VT) {      // [B][H][D][S]
  __shared__ __align__(16) __bf16 lA[16 * LP_];
  int lane = threadIdx.x & 31;
  int wave = threadIdx.x >> 5;
  int m0 = blockIdx.x * 16;
  int o0 = blockIdx.y * 256 + wave * 32;

  for (int idx = threadIdx.x; idx < 16 * (D_ / 8); idx += 256) {
    int row = idx >> 7;
    int c8  = (idx & 127) << 3;
    copy16_g2l(A + (size_t)(m0 + row) * D_ + c8, lA + row * LP_ + c8);
  }
  async_join();

  v8f acc0 = {}, acc1 = {};
  for (int k0 = 0; k0 < D_; k0 += 32) {
    __builtin_prefetch(W + (size_t)o0 * D_ + k0 + 256, 0, 0);
    v16bf a  = load_a16x32(lA + k0, LP_, lane);
    v16bf b0 = load_b32x16(W + (size_t)o0 * D_ + k0, D_, lane);
    v16bf b1 = load_b32x16(W + (size_t)(o0 + 16) * D_ + k0, D_, lane);
    acc0 = wmma_bf16(a, b0, acc0);
    acc1 = wmma_bf16(a, b1, acc1);
  }
  int bb = m0 >> 10;                               // m0 / S_
  int s0 = (m0 & (S_ - 1)) + ((lane >> 4) << 3);
#pragma unroll
  for (int t = 0; t < 2; ++t) {
    int o = o0 + t * 16 + (lane & 15);
    int h = o >> 10;                               // o / D_
    int d = o & (D_ - 1);
    BF8 pk;
    v8f* acc = t ? &acc1 : &acc0;
#pragma unroll
    for (int r = 0; r < 8; ++r) pk.v[r] = f2bf((*acc)[r]);
    *(BF8*)(VT + (size_t)((bb * H_ + h) * D_ + d) * S_ + s0) = pk;
  }
}

// ---------------------------------------------------------------------------
// 4) scores = QK^T * 1/sqrt(L), causal mask, softmax -> probs (bf16)
//    1 block per (b,h,q-tile of 16); Q panel in LDS; 8 waves x 128 keys
// ---------------------------------------------------------------------------
__global__ void attn_scores_softmax(const __bf16* __restrict__ Q,   // [B*S][OQK]
                                    const __bf16* __restrict__ Kb,  // [B*S][OQK]
                                    __bf16* __restrict__ P) {       // [B][H][S][S]
  __shared__ __align__(16) __bf16 lQ[16 * LPQ_];
  __shared__ float red[8][16];
  int lane = threadIdx.x & 31;
  int wave = threadIdx.x >> 5;
  int qt = blockIdx.x & 63;
  int hh = (blockIdx.x >> 6) & 7;
  int bb = blockIdx.x >> 9;
  int q0 = qt * 16;
  int kw = wave * 128;
  const float scale = 0.0625f;     // 1/sqrt(256)

  // stage Q panel: 16 rows x 256 bf16
  const __bf16* qbase = Q + (size_t)(bb * S_ + q0) * OQK + hh * L_;
  for (int idx = threadIdx.x; idx < 16 * (L_ / 8); idx += 256) {
    int row = idx >> 5;                 // idx / 32
    int c8  = (idx & 31) << 3;
    copy16_g2l(qbase + (size_t)row * OQK + c8, lQ + row * LPQ_ + c8);
  }
  async_join();

  v8f acc[8];
#pragma unroll
  for (int t = 0; t < 8; ++t) acc[t] = (v8f){};

  const __bf16* kbase = Kb + (size_t)(bb * S_) * OQK + hh * L_;
  for (int kk = 0; kk < 8; ++kk) {          // L = 256 = 8 * 32
    int l0 = kk * 32;
    v16bf a = load_a16x32(lQ + l0, LPQ_, lane);
#pragma unroll
    for (int t = 0; t < 8; ++t) {
      v16bf b = load_b32x16(kbase + (size_t)(kw + t * 16) * OQK + l0, OQK, lane);
      acc[t] = wmma_bf16(a, b, acc[t]);
    }
  }

  int ncol = lane & 15;
  int hrow = (lane >> 4) << 3;              // 0 or 8
  float rmax[8];
#pragma unroll
  for (int r = 0; r < 8; ++r) rmax[r] = -3.0e38f;

#pragma unroll
  for (int t = 0; t < 8; ++t) {
#pragma unroll
    for (int r = 0; r < 8; ++r) {
      float v = acc[t][r] * scale;
      int key = kw + t * 16 + ncol;
      int qg  = q0 + hrow + r;
      v = (key <= qg) ? v : -3.0e38f;       // causal mask
      acc[t][r] = v;
      rmax[r] = fmaxf(rmax[r], v);
    }
  }
#pragma unroll
  for (int r = 0; r < 8; ++r) {
    float v = rmax[r];
    v = fmaxf(v, __shfl_xor(v, 8, 32));
    v = fmaxf(v, __shfl_xor(v, 4, 32));
    v = fmaxf(v, __shfl_xor(v, 2, 32));
    v = fmaxf(v, __shfl_xor(v, 1, 32));
    rmax[r] = v;
  }
  if (lane == 0 || lane == 16) {
#pragma unroll
    for (int r = 0; r < 8; ++r) red[wave][hrow + r] = rmax[r];
  }
  __syncthreads();
  float gmax[8];
#pragma unroll
  for (int r = 0; r < 8; ++r) {
    float v = red[0][hrow + r];
#pragma unroll
    for (int w = 1; w < 8; ++w) v = fmaxf(v, red[w][hrow + r]);
    gmax[r] = v;
  }
  __syncthreads();

  float rsum[8];
#pragma unroll
  for (int r = 0; r < 8; ++r) rsum[r] = 0.0f;
#pragma unroll
  for (int t = 0; t < 8; ++t) {
#pragma unroll
    for (int r = 0; r < 8; ++r) {
      float p = __expf(acc[t][r] - gmax[r]);
      acc[t][r] = p;
      rsum[r] += p;
    }
  }
#pragma unroll
  for (int r = 0; r < 8; ++r) {
    float v = rsum[r];
    v += __shfl_xor(v, 8, 32);
    v += __shfl_xor(v, 4, 32);
    v += __shfl_xor(v, 2, 32);
    v += __shfl_xor(v, 1, 32);
    rsum[r] = v;
  }
  if (lane == 0 || lane == 16) {
#pragma unroll
    for (int r = 0; r < 8; ++r) red[wave][hrow + r] = rsum[r];
  }
  __syncthreads();
  float inv[8];
#pragma unroll
  for (int r = 0; r < 8; ++r) {
    float v = 0.0f;
#pragma unroll
    for (int w = 0; w < 8; ++w) v += red[w][hrow + r];
    inv[r] = 1.0f / v;
  }

  __bf16* pbase = P + (size_t)((bb * H_ + hh) * S_ + q0 + hrow) * S_ + kw + ncol;
#pragma unroll
  for (int t = 0; t < 8; ++t) {
#pragma unroll
    for (int r = 0; r < 8; ++r) {
      pbase[(size_t)r * S_ + t * 16] = f2bf(acc[t][r] * inv[r]);
    }
  }
}

// ---------------------------------------------------------------------------
// 5) y[b,q,d] = sum_h probs_h @ V_h   (fp32 output)
//    probs panel for each head staged in LDS; wave -> 16x32 output tile
// ---------------------------------------------------------------------------
__global__ void attn_av(const __bf16* __restrict__ P,   // [B][H][S][S]
                        const __bf16* __restrict__ VT,  // [B][H][D][S]
                        float* __restrict__ Y) {        // [B][S][D]
  __shared__ __align__(16) __bf16 lA[16 * LP_];
  int lane = threadIdx.x & 31;
  int wave = threadIdx.x >> 5;
  int m0 = blockIdx.x * 16;
  int bb = m0 >> 10;
  int q0 = m0 & (S_ - 1);
  int d0 = blockIdx.y * 256 + wave * 32;

  v8f acc0 = {}, acc1 = {};
  for (int h = 0; h < H_; ++h) {
    const __bf16* pbase = P + (size_t)((bb * H_ + h) * S_ + q0) * S_;
    __syncthreads();                       // previous panel fully consumed
    for (int idx = threadIdx.x; idx < 16 * (S_ / 8); idx += 256) {
      int row = idx >> 7;
      int c8  = (idx & 127) << 3;
      copy16_g2l(pbase + (size_t)row * S_ + c8, lA + row * LP_ + c8);
    }
    async_join();

    const __bf16* vbase = VT + (size_t)((bb * H_ + h) * D_ + d0) * S_;
    for (int k0 = 0; k0 < S_; k0 += 32) {
      __builtin_prefetch(vbase + k0 + 256, 0, 0);
      v16bf a  = load_a16x32(lA + k0, LP_, lane);
      v16bf b0 = load_b32x16(vbase + k0, S_, lane);
      v16bf b1 = load_b32x16(vbase + (size_t)16 * S_ + k0, S_, lane);
      acc0 = wmma_bf16(a, b0, acc0);
      acc1 = wmma_bf16(a, b1, acc1);
    }
  }
  int hrow = (lane >> 4) << 3;
  int d = d0 + (lane & 15);
#pragma unroll
  for (int r = 0; r < 8; ++r) {
    Y[(size_t)(bb * S_ + q0 + hrow + r) * D_ + d]      = acc0[r];
    Y[(size_t)(bb * S_ + q0 + hrow + r) * D_ + d + 16] = acc1[r];
  }
}

// ---------------------------------------------------------------------------
extern "C" void kernel_launch(void* const* d_in, const int* in_sizes, int n_in,
                              void* d_out, int out_size, void* d_ws, size_t ws_size,
                              hipStream_t stream) {
  const float* x  = (const float*)d_in[0];
  const float* Wq = (const float*)d_in[1];
  const float* Wk = (const float*)d_in[2];
  const float* Wv = (const float*)d_in[3];
  // d_in[4] (mask) unused: causal mask computed analytically.
  float* y = (float*)d_out;

  char* ws = (char*)d_ws;
  size_t off = 0;
  __bf16* xb  = (__bf16*)(ws + off); off += (size_t)M_  * D_ * 2;  //  8 MB
  __bf16* wqb = (__bf16*)(ws + off); off += (size_t)OQK * D_ * 2;  //  4 MB
  __bf16* wkb = (__bf16*)(ws + off); off += (size_t)OQK * D_ * 2;  //  4 MB
  __bf16* wvb = (__bf16*)(ws + off); off += (size_t)OV  * D_ * 2;  // 16 MB
  __bf16* qb  = (__bf16*)(ws + off); off += (size_t)M_  * OQK * 2; // 16 MB
  __bf16* kb  = (__bf16*)(ws + off); off += (size_t)M_  * OQK * 2; // 16 MB
  __bf16* vtb = (__bf16*)(ws + off); off += (size_t)B_ * H_ * D_ * S_ * 2; // 64 MB
  __bf16* pb  = (__bf16*)(ws + off); off += (size_t)B_ * H_ * S_ * S_ * 2; // 64 MB
  (void)ws_size; (void)in_sizes; (void)n_in; (void)out_size;

  // 1) casts
  cast_f32_bf16<<<2048, 256, 0, stream>>>(x,  xb,  M_ * D_);
  cast_f32_bf16<<<2048, 256, 0, stream>>>(Wq, wqb, OQK * D_);
  cast_f32_bf16<<<2048, 256, 0, stream>>>(Wk, wkb, OQK * D_);
  cast_f32_bf16<<<2048, 256, 0, stream>>>(Wv, wvb, OV * D_);

  // 2) Q, K projections: [4096, 2048]
  gemm_xwT<<<dim3(M_ / 16, OQK / 256), 256, 0, stream>>>(xb, wqb, qb, OQK);
  gemm_xwT<<<dim3(M_ / 16, OQK / 256), 256, 0, stream>>>(xb, wkb, kb, OQK);

  // 3) V projection (transposed store): [4096, 8192] -> VT[b][h][d][s]
  gemm_vproj<<<dim3(M_ / 16, OV / 256), 256, 0, stream>>>(xb, wvb, vtb);

  // 4) scores + causal softmax -> probs
  attn_scores_softmax<<<B_ * H_ * (S_ / 16), 256, 0, stream>>>(qb, kb, pb);

  // 5) y = sum_h probs @ V
  attn_av<<<dim3(M_ / 16, D_ / 256), 256, 0, stream>>>(pb, vtb, y);
}
